// Attention_858993459552
// MI455X (gfx1250) — matrix operand order repaired and checked by reading, not compiled, not run on hardware
//
#include <hip/hip_runtime.h>
#include <hip/hip_bf16.h>

typedef __attribute__((ext_vector_type(16))) __bf16 v16bf;
typedef __attribute__((ext_vector_type(8)))  __bf16 v8bf;
typedef __attribute__((ext_vector_type(8)))  float  v8f;
typedef unsigned short u16b;

#define WMMA_BF16(a, b, c) \
  __builtin_amdgcn_wmma_f32_16x16x32_bf16(false, (a), false, (b), (short)0, (c), false, false)

__device__ __forceinline__ u16b f2bf(float f) {
  unsigned u = __float_as_uint(f);
  u += 0x7FFFu + ((u >> 16) & 1u);   // round-to-nearest-even truncate to bf16
  return (u16b)(u >> 16);
}

// A-fragment (16x32 bf16, MxK): lane holds row (lane&15); lanes 0-15 take
// K = {0..7, 16..23}, lanes 16-31 take K = {8..15, 24..31}.
__device__ __forceinline__ v16bf load_a_frag(const u16b* rowptr, int lane) {
  const v8bf* p = (const v8bf*)rowptr;
  int c = (lane >> 4) & 1;
  v8bf lo = p[c];
  v8bf hi = p[c + 2];
  v16bf f;
#pragma unroll
  for (int i = 0; i < 8; ++i) { f[i] = lo[i]; f[8 + i] = hi[i]; }
  return f;
}

// B-fragment (32x16 bf16, KxN): lane holds column (lane&15); lanes 0-15 take
// K = {0..7, 16..23}, lanes 16-31 take K = {8..15, 24..31} (K contiguous in memory).
__device__ __forceinline__ v16bf load_b_frag(const u16b* colptr, int lane) {
  const v8bf* p = (const v8bf*)colptr;
  int c = (lane >> 4) & 1;
  v8bf lo = p[c];
  v8bf hi = p[c + 2];
  v16bf f;
#pragma unroll
  for (int i = 0; i < 8; ++i) { f[i] = lo[i]; f[8 + i] = hi[i]; }
  return f;
}

__global__ void cvt_f32_bf16(const float* __restrict__ src, u16b* __restrict__ dst, int n) {
  for (int i = blockIdx.x * blockDim.x + threadIdx.x; i < n; i += gridDim.x * blockDim.x)
    dst[i] = f2bf(src[i]);
}

// ---------------------------------------------------------------------------
// Kernel 1: QKV projection.  C[m, e] = sum_c X[m,c] * Wqkv[e,c]
// M = 8192 (n*t), N = 3072, K = 1024.  One 64x64 tile per wave.
// Q,K stored [n,h,t,d] bf16; V stored transposed [n,h,d,t] bf16.
// ---------------------------------------------------------------------------
__global__ __launch_bounds__(128) void qkv_gemm(const u16b* __restrict__ xb,
                                                const u16b* __restrict__ wb,
                                                u16b* __restrict__ Qb,
                                                u16b* __restrict__ Kb,
                                                u16b* __restrict__ Vtb) {
  const int lane = threadIdx.x & 31, wv = threadIdx.x >> 5;
  const int lr = lane & 15, hic = (lane >> 4) & 1;
  const int tile = blockIdx.x * 4 + wv;          // 6144 tiles total
  const int m0 = (tile / 48) * 64;
  const int n0 = (tile % 48) * 64;

  v8f acc[4][4];
#pragma unroll
  for (int i = 0; i < 4; ++i)
#pragma unroll
    for (int j = 0; j < 4; ++j)
#pragma unroll
      for (int e = 0; e < 8; ++e) acc[i][j][e] = 0.f;

  for (int k0 = 0; k0 < 1024; k0 += 32) {
    v16bf a0 = load_a_frag(xb + (m0 +  0 + lr) * 1024 + k0, lane);
    v16bf a1 = load_a_frag(xb + (m0 + 16 + lr) * 1024 + k0, lane);
    v16bf a2 = load_a_frag(xb + (m0 + 32 + lr) * 1024 + k0, lane);
    v16bf a3 = load_a_frag(xb + (m0 + 48 + lr) * 1024 + k0, lane);
#pragma unroll
    for (int jn = 0; jn < 4; ++jn) {
      v16bf b = load_b_frag(wb + (n0 + jn * 16 + lr) * 1024 + k0, lane);
      acc[0][jn] = WMMA_BF16(a0, b, acc[0][jn]);
      acc[1][jn] = WMMA_BF16(a1, b, acc[1][jn]);
      acc[2][jn] = WMMA_BF16(a2, b, acc[2][jn]);
      acc[3][jn] = WMMA_BF16(a3, b, acc[3][jn]);
    }
  }

#pragma unroll
  for (int im = 0; im < 4; ++im)
#pragma unroll
    for (int in = 0; in < 4; ++in)
#pragma unroll
      for (int j = 0; j < 8; ++j) {
        int mrow = m0 + im * 16 + j + hic * 8;
        int e = n0 + in * 16 + lr;
        u16b bv = f2bf(acc[im][in][j]);
        int which = e >> 10;
        int r = e & 1023;
        int hh = r >> 6, dd = r & 63;
        int nb = mrow >> 11, tt = mrow & 2047;
        if (which == 0)      Qb[(((nb * 16 + hh) * 2048) + tt) * 64 + dd] = bv;
        else if (which == 1) Kb[(((nb * 16 + hh) * 2048) + tt) * 64 + dd] = bv;
        else                 Vtb[(((nb * 16 + hh) * 64) + dd) * 2048 + tt] = bv;
      }
}

// ---------------------------------------------------------------------------
// Kernel 2: flash attention.  One wave = 16 query rows of one (n,h);
// block = 4 waves = 64 queries.  Online softmax over 2048 keys, 64 per step.
// ---------------------------------------------------------------------------
__global__ __launch_bounds__(128) void attn(const u16b* __restrict__ Qb,
                                            const u16b* __restrict__ Kb,
                                            const u16b* __restrict__ Vtb,
                                            u16b* __restrict__ Ob) {
  const int lane = threadIdx.x & 31, wv = threadIdx.x >> 5;
  const int lr = lane & 15, hic = (lane >> 4) & 1;
  const int nh = blockIdx.y;
  const int n = nh >> 4, h = nh & 15;
  const int q0 = blockIdx.x * 64 + wv * 16;

  const u16b* Qh = Qb + nh * (2048 * 64);
  const u16b* Kh = Kb + nh * (2048 * 64);
  const u16b* Vh = Vtb + nh * (64 * 2048);

  __shared__ __align__(16) u16b pbuf[4][16 * 72];   // padded stride 72 halves
  u16b* myp = pbuf[wv];

  v16bf aq0 = load_a_frag(Qh + (q0 + lr) * 64 + 0, lane);
  v16bf aq1 = load_a_frag(Qh + (q0 + lr) * 64 + 32, lane);

  float m[8], l[8];
  v8f o[4];
#pragma unroll
  for (int j = 0; j < 8; ++j) { m[j] = -1e30f; l[j] = 0.f; }
#pragma unroll
  for (int t4 = 0; t4 < 4; ++t4)
#pragma unroll
    for (int j = 0; j < 8; ++j) o[t4][j] = 0.f;

  const float scale = 0.125f;  // 1/sqrt(64)

  for (int s0 = 0; s0 < 2048; s0 += 64) {
    // --- S = scale * Q K^T  (4 tiles of 16 keys) ---
    v8f st[4];
#pragma unroll
    for (int t4 = 0; t4 < 4; ++t4) {
      v8f c;
#pragma unroll
      for (int j = 0; j < 8; ++j) c[j] = 0.f;
      v16bf b0 = load_b_frag(Kh + (s0 + t4 * 16 + lr) * 64 + 0, lane);
      c = WMMA_BF16(aq0, b0, c);
      v16bf b1 = load_b_frag(Kh + (s0 + t4 * 16 + lr) * 64 + 32, lane);
      c = WMMA_BF16(aq1, b1, c);
#pragma unroll
      for (int j = 0; j < 8; ++j) c[j] *= scale;
      st[t4] = c;
    }
    // --- online softmax: row max (per-lane, then 16-lane butterfly) ---
    float bm[8];
#pragma unroll
    for (int j = 0; j < 8; ++j)
      bm[j] = fmaxf(fmaxf(st[0][j], st[1][j]), fmaxf(st[2][j], st[3][j]));
#pragma unroll
    for (int msk = 1; msk < 16; msk <<= 1)
#pragma unroll
      for (int j = 0; j < 8; ++j) bm[j] = fmaxf(bm[j], __shfl_xor(bm[j], msk));
    float sf[8];
#pragma unroll
    for (int j = 0; j < 8; ++j) {
      float mn = fmaxf(m[j], bm[j]);
      sf[j] = __expf(m[j] - mn);
      m[j] = mn;
      l[j] *= sf[j];
    }
#pragma unroll
    for (int t4 = 0; t4 < 4; ++t4)
#pragma unroll
      for (int j = 0; j < 8; ++j) o[t4][j] *= sf[j];
    // --- P = exp(S - m), row sums ---
    float rs[8];
#pragma unroll
    for (int j = 0; j < 8; ++j) rs[j] = 0.f;
#pragma unroll
    for (int t4 = 0; t4 < 4; ++t4)
#pragma unroll
      for (int j = 0; j < 8; ++j) {
        float p = __expf(st[t4][j] - m[j]);
        st[t4][j] = p;
        rs[j] += p;
      }
#pragma unroll
    for (int msk = 1; msk < 16; msk <<= 1)
#pragma unroll
      for (int j = 0; j < 8; ++j) rs[j] += __shfl_xor(rs[j], msk);
#pragma unroll
    for (int j = 0; j < 8; ++j) l[j] += rs[j];
    // --- C-layout P -> bf16 via LDS -> A-fragment layout ---
#pragma unroll
    for (int t4 = 0; t4 < 4; ++t4)
#pragma unroll
      for (int j = 0; j < 8; ++j)
        myp[(j + hic * 8) * 72 + t4 * 16 + lr] = f2bf(st[t4][j]);
    v16bf pa0 = load_a_frag(myp + lr * 72 + 0, lane);
    v16bf pa1 = load_a_frag(myp + lr * 72 + 32, lane);
    // --- O += P V  (V stored transposed [d, t]: contiguous B-frags) ---
#pragma unroll
    for (int dn = 0; dn < 4; ++dn) {
      v16bf bv0 = load_b_frag(Vh + (dn * 16 + lr) * 2048 + s0 + 0, lane);
      o[dn] = WMMA_BF16(pa0, bv0, o[dn]);
      v16bf bv1 = load_b_frag(Vh + (dn * 16 + lr) * 2048 + s0 + 32, lane);
      o[dn] = WMMA_BF16(pa1, bv1, o[dn]);
    }
  }

  // normalize + store to merged-head layout [n, t, h*d] as bf16
#pragma unroll
  for (int dn = 0; dn < 4; ++dn)
#pragma unroll
    for (int j = 0; j < 8; ++j) {
      int row = j + hic * 8;
      int tt = q0 + row;
      int col = h * 64 + dn * 16 + lr;
      float val = o[dn][j] / l[j];
      Ob[(n * 2048 + tt) * 1024 + col] = f2bf(val);
    }
}

// ---------------------------------------------------------------------------
// Kernel 3: output projection.  out[m,e] = sum_c O[m,c]*Wout[e,c] + b[e]
// ---------------------------------------------------------------------------
__global__ __launch_bounds__(128) void out_gemm(const u16b* __restrict__ Ob,
                                                const u16b* __restrict__ wb,
                                                const float* __restrict__ bias,
                                                float* __restrict__ out) {
  const int lane = threadIdx.x & 31, wv = threadIdx.x >> 5;
  const int lr = lane & 15, hic = (lane >> 4) & 1;
  const int tile = blockIdx.x * 4 + wv;          // 2048 tiles
  const int m0 = (tile / 16) * 64;
  const int n0 = (tile % 16) * 64;

  v8f acc[4][4];
#pragma unroll
  for (int i = 0; i < 4; ++i)
#pragma unroll
    for (int j = 0; j < 4; ++j)
#pragma unroll
      for (int e = 0; e < 8; ++e) acc[i][j][e] = 0.f;

  for (int k0 = 0; k0 < 1024; k0 += 32) {
    v16bf a0 = load_a_frag(Ob + (m0 +  0 + lr) * 1024 + k0, lane);
    v16bf a1 = load_a_frag(Ob + (m0 + 16 + lr) * 1024 + k0, lane);
    v16bf a2 = load_a_frag(Ob + (m0 + 32 + lr) * 1024 + k0, lane);
    v16bf a3 = load_a_frag(Ob + (m0 + 48 + lr) * 1024 + k0, lane);
#pragma unroll
    for (int jn = 0; jn < 4; ++jn) {
      v16bf b = load_b_frag(wb + (n0 + jn * 16 + lr) * 1024 + k0, lane);
      acc[0][jn] = WMMA_BF16(a0, b, acc[0][jn]);
      acc[1][jn] = WMMA_BF16(a1, b, acc[1][jn]);
      acc[2][jn] = WMMA_BF16(a2, b, acc[2][jn]);
      acc[3][jn] = WMMA_BF16(a3, b, acc[3][jn]);
    }
  }

#pragma unroll
  for (int im = 0; im < 4; ++im)
#pragma unroll
    for (int in = 0; in < 4; ++in)
#pragma unroll
      for (int j = 0; j < 8; ++j) {
        int mrow = m0 + im * 16 + j + hic * 8;
        int e = n0 + in * 16 + lr;
        out[mrow * 1024 + e] = acc[im][in][j] + bias[e];
      }
}

// ---------------------------------------------------------------------------
extern "C" void kernel_launch(void* const* d_in, const int* in_sizes, int n_in,
                              void* d_out, int out_size, void* d_ws, size_t ws_size,
                              hipStream_t stream) {
  (void)in_sizes; (void)n_in; (void)out_size; (void)ws_size;
  const float* x     = (const float*)d_in[0];  // [4, 2048, 1024]
  const float* w_qkv = (const float*)d_in[1];  // [3072, 1024]
  const float* w_out = (const float*)d_in[2];  // [1024, 1024]
  const float* b_out = (const float*)d_in[3];  // [1024]
  float* out = (float*)d_out;                  // [4, 2048, 1024] fp32

  unsigned char* ws = (unsigned char*)d_ws;
  u16b* xb    = (u16b*)(ws + 0);          // 16,777,216 B
  u16b* wqkvb = (u16b*)(ws + 16777216);   //  6,291,456 B
  u16b* woutb = (u16b*)(ws + 23068672);   //  2,097,152 B
  u16b* Qb    = (u16b*)(ws + 25165824);   // 16,777,216 B  [n,h,t,d]
  u16b* Kb    = (u16b*)(ws + 41943040);   // 16,777,216 B  [n,h,t,d]
  u16b* Vtb   = (u16b*)(ws + 58720256);   // 16,777,216 B  [n,h,d,t]
  u16b* Ob    = (u16b*)(ws + 75497472);   // 16,777,216 B  [n,t,h*d]

  cvt_f32_bf16<<<2048, 256, 0, stream>>>(x, xb, 4 * 2048 * 1024);
  cvt_f32_bf16<<<1024, 256, 0, stream>>>(w_qkv, wqkvb, 3072 * 1024);
  cvt_f32_bf16<<<512, 256, 0, stream>>>(w_out, woutb, 1024 * 1024);

  qkv_gemm<<<1536, 128, 0, stream>>>(xb, wqkvb, Qb, Kb, Vtb);
  attn<<<dim3(32, 64), 128, 0, stream>>>(Qb, Kb, Vtb, Ob);
  out_gemm<<<512, 128, 0, stream>>>(Ob, woutb, b_out, out);
}
